// sheafHyperGCN_62998580297953
// MI455X (gfx1250) — matrix operand order repaired and checked by reading, not compiled, not run on hardware
//
#include <hip/hip_runtime.h>

typedef __attribute__((ext_vector_type(16))) _Float16 v16h;
typedef __attribute__((ext_vector_type(8)))  _Float16 v8h;
typedef __attribute__((ext_vector_type(8)))  float    v8f;

#define DEV_INLINE __device__ __forceinline__

// Problem constants (from reference)
constexpr int V    = 50000;
constexpr int E    = 25000;
constexpr int Epad = 25008;          // E rounded up to 16 (lift M-tile padding)
constexpr int NNZ  = 200000;
constexpr int F_IN = 128;
constexpr int HID  = 256;
constexpr int D    = 4;
constexpr int H1   = 16;
constexpr int C    = 40;
constexpr int DH   = D * HID;        // 1024
constexpr int Cpad = 48;             // C rounded up to 16
constexpr int K2p  = 32;             // conv2 K=16 padded to 32

// ---------------------------------------------------------------------------
// WMMA 16x16x32 f16->f32 (CDNA5 ISA §7.12)
// ---------------------------------------------------------------------------
DEV_INLINE v8f wmma_f32(v16h a, v16h b, v8f c) {
  return __builtin_amdgcn_wmma_f32_16x16x32_f16(false, a, false, b, (short)0, c,
                                                false, false);
}

DEV_INLINE void pack8(v16h& out, int base, float4 a, float4 b) {
  out[base + 0] = (_Float16)a.x; out[base + 1] = (_Float16)a.y;
  out[base + 2] = (_Float16)a.z; out[base + 3] = (_Float16)a.w;
  out[base + 4] = (_Float16)b.x; out[base + 5] = (_Float16)b.y;
  out[base + 6] = (_Float16)b.z; out[base + 7] = (_Float16)b.w;
}

DEV_INLINE float4 f4lap(float4 h, float4 q, float iv) {
  return make_float4(h.x - iv * q.x, h.y - iv * q.y, h.z - iv * q.z, h.w - iv * q.w);
}

// A fragment 16x32 (ISA layout: lanes 0-15 row=lane, K={0..7,16..23}; lanes
// 16-31 K={8..15,24..31}).  Four global_load_b128 per lane; row clamped (no
// divergence) — callers guarantee clamped/pad rows are never consumed.
DEV_INLINE v16h fragA(const float* __restrict__ A, int ld, int row0, int k0,
                      int lane, int mclamp) {
  const int half = lane >> 4, r = lane & 15;
  int m = row0 + r;
  m = (m < mclamp) ? m : (mclamp - 1);
  const float* p = A + (size_t)m * ld + k0 + half * 8;
  float4 a0 = *(const float4*)(p);
  float4 a1 = *(const float4*)(p + 4);
  float4 a2 = *(const float4*)(p + 16);
  float4 a3 = *(const float4*)(p + 20);
  v16h out;
  pack8(out, 0, a0, a1);
  pack8(out, 8, a2, a3);
  return out;
}

// A fragment with fused sheaf Laplacian: L[m,k] = Hsrc[m,k] - inv_v[m>>2]*acc[m,k]
DEV_INLINE v16h fragA_lap(const float* __restrict__ Hsrc, const float* __restrict__ acc,
                          const float* __restrict__ invv, int ld, int row0, int k0,
                          int lane) {
  const int half = lane >> 4, r = lane & 15;
  const int m = row0 + r;
  const float iv = invv[m >> 2];
  const float* ph = Hsrc + (size_t)m * ld + k0 + half * 8;
  const float* pa = acc  + (size_t)m * ld + k0 + half * 8;
  float4 h0 = *(const float4*)(ph),      h1 = *(const float4*)(ph + 4);
  float4 h2 = *(const float4*)(ph + 16), h3 = *(const float4*)(ph + 20);
  float4 q0 = *(const float4*)(pa),      q1 = *(const float4*)(pa + 4);
  float4 q2 = *(const float4*)(pa + 16), q3 = *(const float4*)(pa + 20);
  v16h out;
  pack8(out, 0, f4lap(h0, q0, iv), f4lap(h1, q1, iv));
  pack8(out, 8, f4lap(h2, q2, iv), f4lap(h3, q3, iv));
  return out;
}

// A fragment for K=16 (conv2): valid K occupies exactly elements 0..7 of the
// fragment (k = half*8 + e); elements 8..15 (k=16..31) are zero to match the
// zero-padded K of the f16 weight copy.
DEV_INLINE v16h fragA_lap16(const float* __restrict__ Hsrc, const float* __restrict__ acc,
                            const float* __restrict__ invv, int row0, int lane) {
  const int half = lane >> 4, r = lane & 15;
  const int m = row0 + r;
  const float iv = invv[m >> 2];
  const float* ph = Hsrc + (size_t)m * H1 + half * 8;
  const float* pa = acc  + (size_t)m * H1 + half * 8;
  float4 h0 = *(const float4*)(ph), h1 = *(const float4*)(ph + 4);
  float4 q0 = *(const float4*)(pa), q1 = *(const float4*)(pa + 4);
  v16h out;
  pack8(out, 0, f4lap(h0, q0, iv), f4lap(h1, q1, iv));
#pragma unroll
  for (int e = 8; e < 16; ++e) out[e] = (_Float16)0.0f;
  return out;
}

// B fragment from pre-transposed, zero-padded f16 weights Bt[Npad, Kpad]:
// lane's K-values are {half*8..+7, 16+half*8..+7} -> two global_load_b128, no guards.
DEV_INLINE v16h fragB_t(const _Float16* __restrict__ Bt, int ldk, int k0, int col0,
                        int lane) {
  const int half = lane >> 4, cidx = lane & 15;
  const _Float16* p = Bt + (size_t)(col0 + cidx) * ldk + k0 + half * 8;
  v8h b0 = *(const v8h*)(p);
  v8h b1 = *(const v8h*)(p + 16);
  v16h out;
#pragma unroll
  for (int e = 0; e < 8; ++e) { out[e] = b0[e]; out[e + 8] = b1[e]; }
  return out;
}

// C store (ISA layout: VGPR i -> row = row0 + 8*half + i, col = col0 + lane%16).
// Rows are unguarded (outputs padded); only the N=40 column edge is guarded,
// and only after all WMMAs for the tile have issued.
DEV_INLINE void storeC(float* __restrict__ Y, int ldn, int row0, int col0, int lane,
                       v8f c, const float* __restrict__ bias, int nmax, bool relu) {
  const int half = lane >> 4, cidx = lane & 15;
  const int col = col0 + cidx;
  if (col >= nmax) return;
  const float b = bias ? bias[col] : 0.0f;
#pragma unroll
  for (int i = 0; i < 8; ++i) {
    float x = c[i] + b;
    if (relu) x = fmaxf(x, 0.0f);
    Y[(size_t)(row0 + half * 8 + i) * ldn + col] = x;
  }
}

// ---------------------------------------------------------------------------
// Weight pre-pass: W[K,N] f32 -> Wt[Npad,Kpad] f16, transposed, zero-padded
// ---------------------------------------------------------------------------
__global__ void k_cvt_wt(const float* __restrict__ W, _Float16* __restrict__ Wt,
                         int K, int N, int Kpad, int Npad) {
  int i = blockIdx.x * blockDim.x + threadIdx.x;
  if (i >= Kpad * Npad) return;
  int n = i / Kpad, k = i % Kpad;
  float v = (n < N && k < K) ? W[(size_t)k * N + n] : 0.0f;
  Wt[i] = (_Float16)v;
}

// ---------------------------------------------------------------------------
// Kernel 1: stalk lift  Y[M,1024] = X[M,128] @ Wlin + blin  (16x64 per wave)
// ---------------------------------------------------------------------------
__global__ void k_lift(const float* __restrict__ X, const _Float16* __restrict__ Wt,
                       const float* __restrict__ bias, float* __restrict__ Y, int M) {
  const int lane = threadIdx.x & 31;
  const int wid  = blockIdx.x * (blockDim.x >> 5) + (threadIdx.x >> 5);
  const int tilesM = (M + 15) >> 4;
  const int strips = DH / 64;
  if (wid >= tilesM * strips) return;
  const int tm = wid / strips, tn = wid % strips;
  const int row0 = tm << 4, col0 = tn * 64;
  v8f acc[4] = {};
  for (int k0 = 0; k0 < F_IN; k0 += 32) {
    v16h a = fragA(X, F_IN, row0, k0, lane, M);
#pragma unroll
    for (int t = 0; t < 4; ++t) {
      v16h b = fragB_t(Wt, F_IN, k0, col0 + t * 16, lane);
      acc[t] = wmma_f32(a, b, acc[t]);
    }
  }
#pragma unroll
  for (int t = 0; t < 4; ++t)
    storeC(Y, DH, row0, col0 + t * 16, lane, acc[t], bias, DH, false);
}

// ---------------------------------------------------------------------------
// Kernel 2: restriction-map scalars (wave per incidence, 32-lane dot over 256)
// ---------------------------------------------------------------------------
__global__ void k_sheaf_s(const float* __restrict__ Hn, const float* __restrict__ He,
                          const int* __restrict__ nidx, const int* __restrict__ eidx,
                          const float* __restrict__ w1, const float* __restrict__ w2,
                          const float* __restrict__ bs, float* __restrict__ s) {
  const int lane = threadIdx.x & 31;
  const int n = blockIdx.x * (blockDim.x >> 5) + (threadIdx.x >> 5);
  if (n >= NNZ) return;
  const float* pv = Hn + (size_t)nidx[n] * DH;
  const float* pe = He + (size_t)eidx[n] * DH;
#pragma unroll
  for (int d = 0; d < D; ++d) {
    float acc = 0.0f;
    for (int j = lane; j < HID; j += 32)
      acc += pv[d * HID + j] * w1[j * D + d] + pe[d * HID + j] * w2[j * D + d];
#pragma unroll
    for (int off = 16; off > 0; off >>= 1)
      acc += __shfl_xor(acc, off, 32);
    if (lane == 0) s[(size_t)n * D + d] = tanhf(acc + bs[d]);
  }
}

// ---------------------------------------------------------------------------
// Degrees + inversion
// ---------------------------------------------------------------------------
__global__ void k_degree(const int* __restrict__ nidx, const int* __restrict__ eidx,
                         float* __restrict__ deg_v, float* __restrict__ deg_e) {
  int i = blockIdx.x * blockDim.x + threadIdx.x;
  if (i >= NNZ) return;
  atomicAdd(&deg_v[nidx[i]], 1.0f);
  atomicAdd(&deg_e[eidx[i]], 1.0f);
}

__global__ void k_invert(float* __restrict__ x, int n) {
  int i = blockIdx.x * blockDim.x + threadIdx.x;
  if (i < n) { float v = x[i]; x[i] = (v > 0.0f) ? 1.0f / v : 0.0f; }
}

// ---------------------------------------------------------------------------
// Scatter: dst[sidx[n], d, :] += s[n,d] * src[gidx[n], d, :]  (wave/incidence)
// Atomic targets are L2-resident (m: 102 MB << 192 MB L2).
// ---------------------------------------------------------------------------
__global__ void k_scatter(const float* __restrict__ src, const float* __restrict__ s,
                          const int* __restrict__ gidx, const int* __restrict__ sidx,
                          float* __restrict__ dst, int c) {
  const int lane = threadIdx.x & 31;
  const int n = blockIdx.x * (blockDim.x >> 5) + (threadIdx.x >> 5);
  if (n >= NNZ) return;
  const int g = gidx[n], sc = sidx[n];
  const int len = D * c;
  const float* ps = src + (size_t)g * len;
  float* pd = dst + (size_t)sc * len;
  const float* pw = s + (size_t)n * D;
  __builtin_prefetch(ps, 0, 1);     // global_prefetch_b8 on the gathered row
  for (int idx = lane; idx < len; idx += 32) {
    float w = pw[idx / c];
    atomicAdd(&pd[idx], w * ps[idx]);
  }
}

__global__ void k_rowscale(float* __restrict__ x, const float* __restrict__ inv,
                           int rowlen, int n) {
  int i = blockIdx.x * blockDim.x + threadIdx.x;
  if (i < n) x[i] *= inv[i / rowlen];
}

// ---------------------------------------------------------------------------
// Conv layer 1: H = relu((Hn - inv_v*a) @ W0 + b0), M=200000, K=256, N=16
// ---------------------------------------------------------------------------
__global__ void k_conv1(const float* __restrict__ Hn, const float* __restrict__ a,
                        const float* __restrict__ invv, const _Float16* __restrict__ W0t,
                        const float* __restrict__ b0, float* __restrict__ H) {
  const int lane = threadIdx.x & 31;
  const int wid  = blockIdx.x * (blockDim.x >> 5) + (threadIdx.x >> 5);
  const int tilesM = (V * D) >> 4;         // 12500, exact
  if (wid >= tilesM) return;
  const int row0 = wid << 4;
  v8f acc = {};
  for (int k0 = 0; k0 < HID; k0 += 32) {
    v16h af = fragA_lap(Hn, a, invv, HID, row0, k0, lane);
    v16h bf = fragB_t(W0t, HID, k0, 0, lane);
    acc = wmma_f32(af, bf, acc);
  }
  storeC(H, H1, row0, 0, lane, acc, b0, H1, true);
}

// ---------------------------------------------------------------------------
// Conv layer 2: H2 = relu((H - inv_v*a2) @ W1 + b1), M=200000, K=16(pad32), N=40
// ---------------------------------------------------------------------------
__global__ void k_conv2(const float* __restrict__ H, const float* __restrict__ a2,
                        const float* __restrict__ invv, const _Float16* __restrict__ W1t,
                        const float* __restrict__ b1, float* __restrict__ H2) {
  const int lane = threadIdx.x & 31;
  const int wid  = blockIdx.x * (blockDim.x >> 5) + (threadIdx.x >> 5);
  const int tilesM = (V * D) >> 4;
  const int ntiles = Cpad / 16;            // 3
  if (wid >= tilesM * ntiles) return;
  const int tm = wid / ntiles, tn = wid % ntiles;
  const int row0 = tm << 4, col0 = tn << 4;
  v16h af = fragA_lap16(H, a2, invv, row0, lane);
  v16h bf = fragB_t(W1t, K2p, 0, col0, lane);
  v8f acc = {};
  acc = wmma_f32(af, bf, acc);
  storeC(H2, C, row0, col0, lane, acc, b1, C, true);
}

// ---------------------------------------------------------------------------
// Classifier: out = H2.reshape(V,160) @ W2, M=V, K=160, N=40
// ---------------------------------------------------------------------------
__global__ void k_out(const float* __restrict__ H2, const _Float16* __restrict__ W2t,
                      float* __restrict__ out) {
  const int lane = threadIdx.x & 31;
  const int wid  = blockIdx.x * (blockDim.x >> 5) + (threadIdx.x >> 5);
  const int tilesM = V >> 4;               // 3125, exact
  const int ntiles = Cpad / 16;
  if (wid >= tilesM * ntiles) return;
  const int tm = wid / ntiles, tn = wid % ntiles;
  const int row0 = tm << 4, col0 = tn << 4;
  const int K = D * C;                     // 160
  v8f acc = {};
  for (int k0 = 0; k0 < K; k0 += 32) {
    v16h af = fragA(H2, K, row0, k0, lane, V);
    v16h bf = fragB_t(W2t, K, k0, col0, lane);
    acc = wmma_f32(af, bf, acc);
  }
  storeC(out, C, row0, col0, lane, acc, nullptr, C, false);
}

// ---------------------------------------------------------------------------
// Host launcher
// ---------------------------------------------------------------------------
extern "C" void kernel_launch(void* const* d_in, const int* in_sizes, int n_in,
                              void* d_out, int out_size, void* d_ws, size_t ws_size,
                              hipStream_t stream) {
  (void)in_sizes; (void)n_in; (void)out_size; (void)ws_size;
  const float* x     = (const float*)d_in[0];
  const float* eattr = (const float*)d_in[1];
  const int*   nidx  = (const int*)d_in[2];
  const int*   eidx  = (const int*)d_in[3];
  const float* Wlin  = (const float*)d_in[4];
  const float* blin  = (const float*)d_in[5];
  const float* w1    = (const float*)d_in[6];
  const float* w2    = (const float*)d_in[7];
  const float* bs    = (const float*)d_in[8];
  const float* W0    = (const float*)d_in[9];
  const float* b0    = (const float*)d_in[10];
  const float* W1    = (const float*)d_in[11];
  const float* b1    = (const float*)d_in[12];
  const float* W2    = (const float*)d_in[13];
  float* out = (float*)d_out;

  // Workspace carve-up (lifetime-aware reuse; peak ~530 MB)
  char* ws = (char*)d_ws;
  size_t off = 0;
  auto take = [&](size_t bytes) -> char* {
    char* p = ws + off;
    off = (off + bytes + 511) & ~(size_t)511;
    return p;
  };
  float* Hn   = (float*)take((size_t)V * DH * sizeof(float));        // [V,1024]
  float* He   = (float*)take((size_t)Epad * DH * sizeof(float));     // padded; reused as m/m2
  float* Aagg = (float*)take((size_t)V * DH * sizeof(float));        // a; reused as a2
  float* Sbuf = (float*)take((size_t)NNZ * D * sizeof(float));
  float* invE = (float*)take((size_t)E * sizeof(float));
  float* invV = (float*)take((size_t)V * sizeof(float));
  float* Hmid = (float*)take((size_t)V * D * H1 * sizeof(float));    // [200000,16]
  _Float16* WlinT = (_Float16*)take((size_t)DH * F_IN * sizeof(_Float16));  // [1024,128]
  _Float16* W0t   = (_Float16*)take((size_t)H1 * HID * sizeof(_Float16));   // [16,256]
  _Float16* W1t   = (_Float16*)take((size_t)Cpad * K2p * sizeof(_Float16)); // [48,32]
  _Float16* W2t   = (_Float16*)take((size_t)Cpad * (D * C) * sizeof(_Float16)); // [48,160]
  float* M1 = He;     // [E, D*HID]
  float* M2 = He;     // [E, D*H1]
  float* A2 = Aagg;   // [V, D*H1]
  float* H2 = Hn;     // [V*D, C]  (Hn dead after k_conv1)

  const int TB  = 256;           // 8 wave32 per block
  const int WPB = TB / 32;
  auto wblocks = [&](long long waves) { return (int)((waves + WPB - 1) / WPB); };
  auto tblocks = [&](long long thr)   { return (int)((thr + TB - 1) / TB); };

  // 0) weight pre-pass: transposed, zero-padded f16 copies
  k_cvt_wt<<<tblocks((long long)DH * F_IN), TB, 0, stream>>>(Wlin, WlinT, F_IN, DH, F_IN, DH);
  k_cvt_wt<<<tblocks((long long)H1 * HID), TB, 0, stream>>>(W0, W0t, HID, H1, HID, H1);
  k_cvt_wt<<<tblocks((long long)Cpad * K2p), TB, 0, stream>>>(W1, W1t, H1, C, K2p, Cpad);
  k_cvt_wt<<<tblocks((long long)Cpad * D * C), TB, 0, stream>>>(W2, W2t, D * C, C, D * C, Cpad);

  // 1) stalk lift GEMMs (WMMA)
  k_lift<<<wblocks((long long)((V + 15) / 16) * (DH / 64)), TB, 0, stream>>>(
      x, WlinT, blin, Hn, V);
  k_lift<<<wblocks((long long)((E + 15) / 16) * (DH / 64)), TB, 0, stream>>>(
      eattr, WlinT, blin, He, E);

  // 2) degree normalization D^{-1}
  hipMemsetAsync(invE, 0, (size_t)E * sizeof(float), stream);
  hipMemsetAsync(invV, 0, (size_t)V * sizeof(float), stream);
  k_degree<<<tblocks(NNZ), TB, 0, stream>>>(nidx, eidx, invV, invE);
  k_invert<<<tblocks(E), TB, 0, stream>>>(invE, E);
  k_invert<<<tblocks(V), TB, 0, stream>>>(invV, V);

  // 3) diagonal restriction-map scalars
  k_sheaf_s<<<wblocks(NNZ), TB, 0, stream>>>(Hn, He, nidx, eidx, w1, w2, bs, Sbuf);

  // 4) sheaf Laplacian pass 1 (c = HID) + conv layer 1
  hipMemsetAsync(M1, 0, (size_t)E * DH * sizeof(float), stream);
  k_scatter<<<wblocks(NNZ), TB, 0, stream>>>(Hn, Sbuf, nidx, eidx, M1, HID);
  k_rowscale<<<tblocks((long long)E * DH), TB, 0, stream>>>(M1, invE, DH, E * DH);
  hipMemsetAsync(Aagg, 0, (size_t)V * DH * sizeof(float), stream);
  k_scatter<<<wblocks(NNZ), TB, 0, stream>>>(M1, Sbuf, eidx, nidx, Aagg, HID);
  k_conv1<<<wblocks((long long)(V * D) / 16), TB, 0, stream>>>(Hn, Aagg, invV, W0t,
                                                               b0, Hmid);

  // 5) sheaf Laplacian pass 2 (c = H1) + conv layer 2
  hipMemsetAsync(M2, 0, (size_t)E * D * H1 * sizeof(float), stream);
  k_scatter<<<wblocks(NNZ), TB, 0, stream>>>(Hmid, Sbuf, nidx, eidx, M2, H1);
  k_rowscale<<<tblocks((long long)E * D * H1), TB, 0, stream>>>(M2, invE, D * H1,
                                                                E * D * H1);
  hipMemsetAsync(A2, 0, (size_t)V * D * H1 * sizeof(float), stream);
  k_scatter<<<wblocks(NNZ), TB, 0, stream>>>(M2, Sbuf, eidx, nidx, A2, H1);
  k_conv2<<<wblocks((long long)((V * D) / 16) * (Cpad / 16)), TB, 0, stream>>>(
      Hmid, A2, invV, W1t, b1, H2);

  // 6) classifier
  k_out<<<wblocks((long long)(V / 16) * (Cpad / 16)), TB, 0, stream>>>(H2, W2t, out);
}
